// TransformerBlock_15693810499845
// MI455X (gfx1250) — compile-verified, matrix operand order checked by reading, not run on hardware
//
#include <hip/hip_runtime.h>
#include <math.h>

typedef __attribute__((ext_vector_type(16))) _Float16 v16h;
typedef __attribute__((ext_vector_type(8)))  float    v8f;

#define T_TOK 1568
#define Dm    768
#define HDk   64
#define Hh    12
#define FFk   3072
#define Ee    8
#define NTIL  98          // ceil(1568/16)  (vproj tiles)
#define MTM   32          // MoE M-tile
#define NT32  49          // ceil(1568/32)  (moe tiles)
#define EPSL  1e-5f

// ---------- WMMA fragment helpers (CDNA5 16x16x32 f16, wave32) ----------
// A (16xK tile, row-major f16 in LDS): lane L -> row m=L&15; K-halves:
//   elems 0..7  -> K = k0 + 8*(L>>4) + j ; elems 8..15 -> K = k0 + 16 + 8*(L>>4) + (j-8)
// Each 8-half group is 16B aligned -> lowers to ds_load_b128 pairs.
__device__ __forceinline__ v16h load_a_lds(const _Float16* As, int lda, int k0) {
  int lane = threadIdx.x & 31;
  const _Float16* row = As + (lane & 15) * lda + k0 + ((lane >> 4) << 3);
  v16h a;
#pragma unroll
  for (int j = 0; j < 8; ++j) a[j] = row[j];
#pragma unroll
  for (int j = 0; j < 8; ++j) a[8 + j] = row[16 + j];
  return a;
}

// B (KxN row-major f32 in global, converted to f16): lane L -> col n=n0+(L&15),
// same K striping as A.
__device__ __forceinline__ v16h load_b_f32(const float* B, int ldb, int k0, int n0) {
  int lane = threadIdx.x & 31;
  const float* p = B + (size_t)(k0 + ((lane >> 4) << 3)) * ldb + n0 + (lane & 15);
  v16h b;
#pragma unroll
  for (int j = 0; j < 8; ++j) b[j] = (_Float16)p[(size_t)j * ldb];
#pragma unroll
  for (int j = 0; j < 8; ++j) b[8 + j] = (_Float16)p[(size_t)(16 + j) * ldb];
  return b;
}

__device__ __forceinline__ v8f wmma16(v16h a, v16h b, v8f c) {
  return __builtin_amdgcn_wmma_f32_16x16x32_f16(false, a, false, b, (short)0, c,
                                                false, false);
}

// ---------- kernel 0: zero routing counters ----------
__global__ void init_counts_kernel(int* counts) {
  if (threadIdx.x < Ee) counts[threadIdx.x] = 0;
}

// ---------- kernel 1: LN1 -> f16 ----------
__global__ void ln1_kernel(const float* __restrict__ x, const float* __restrict__ g,
                           const float* __restrict__ bt, _Float16* __restrict__ outh) {
  __shared__ float ws1[8], ws2[8];
  int tok = blockIdx.x, tid = threadIdx.x;
  const float* row = x + (size_t)tok * Dm;
  float v[3], s = 0.f, s2 = 0.f;
#pragma unroll
  for (int i = 0; i < 3; ++i) { float t = row[tid + i * 256]; v[i] = t; s += t; s2 += t * t; }
#pragma unroll
  for (int off = 16; off; off >>= 1) { s += __shfl_xor(s, off, 32); s2 += __shfl_xor(s2, off, 32); }
  int wv = tid >> 5, lane = tid & 31;
  if (lane == 0) { ws1[wv] = s; ws2[wv] = s2; }
  __syncthreads();
  if (tid == 0) {
    float a = 0.f, b = 0.f;
#pragma unroll
    for (int i = 0; i < 8; ++i) { a += ws1[i]; b += ws2[i]; }
    float mu = a / Dm;
    ws1[0] = mu; ws2[0] = rsqrtf(b / Dm - mu * mu + EPSL);
  }
  __syncthreads();
  float mu = ws1[0], rstd = ws2[0];
#pragma unroll
  for (int i = 0; i < 3; ++i) {
    int c = tid + i * 256;
    outh[(size_t)tok * Dm + c] = (_Float16)((v[i] - mu) * rstd * g[c] + bt[c]);
  }
}

// ---------- kernel 2: v = LN1 @ Wv + bv ; x1 = x + tile(v, 12 heads) ----------
__global__ void vproj_kernel(const _Float16* __restrict__ ln1, const float* __restrict__ Wv,
                             const float* __restrict__ bv, const float* __restrict__ x,
                             float* __restrict__ x1) {
  const int LDA = Dm + 8;
  __shared__ _Float16 As[16 * (Dm + 8)];
  int m0 = blockIdx.x * 16, tid = threadIdx.x;   // 128 threads = 4 waves
  for (int i = tid; i < 16 * Dm; i += 128) {
    int r = i / Dm, c = i - r * Dm;
    int tok = m0 + r;
    As[r * LDA + c] = (tok < T_TOK) ? ln1[(size_t)tok * Dm + c] : (_Float16)0.f;
  }
  __syncthreads();
  int wv = tid >> 5, lane = tid & 31;
  int n0 = wv * 16;                               // HD=64 -> 4 N-tiles, 1 per wave
  v8f acc = {};
  for (int k0 = 0; k0 < Dm; k0 += 32)
    acc = wmma16(load_a_lds(As, LDA, k0), load_b_f32(Wv, HDk, k0, n0), acc);
  int n = n0 + (lane & 15);
  int mh = (lane >> 4) << 3;
  float bvn = bv[n];
#pragma unroll
  for (int r = 0; r < 8; ++r) {
    int tok = m0 + r + mh;
    if (tok < T_TOK) {
      float vval = acc[r] + bvn;
      size_t base = (size_t)tok * Dm;
#pragma unroll
      for (int h = 0; h < Hh; ++h) {
        int c = h * HDk + n;
        x1[base + c] = x[base + c] + vval;
      }
    }
  }
}

// ---------- kernel 3: LN2 -> f16, top-1 routing via atomic scatter ----------
__global__ void route_kernel(const float* __restrict__ x1, const float* __restrict__ g,
                             const float* __restrict__ bt, const float* __restrict__ Wg,
                             const float* __restrict__ bg, _Float16* __restrict__ h2,
                             int* __restrict__ counts, int* __restrict__ idxlist) {
  __shared__ float sh[Dm];
  __shared__ float ws1[8], ws2[8], logits[8];
  int tok = blockIdx.x, tid = threadIdx.x;
  const float* row = x1 + (size_t)tok * Dm;
  float v[3], s = 0.f, s2 = 0.f;
#pragma unroll
  for (int i = 0; i < 3; ++i) { float t = row[tid + i * 256]; v[i] = t; s += t; s2 += t * t; }
#pragma unroll
  for (int off = 16; off; off >>= 1) { s += __shfl_xor(s, off, 32); s2 += __shfl_xor(s2, off, 32); }
  int wv = tid >> 5, lane = tid & 31;
  if (lane == 0) { ws1[wv] = s; ws2[wv] = s2; }
  __syncthreads();
  if (tid == 0) {
    float a = 0.f, b = 0.f;
#pragma unroll
    for (int i = 0; i < 8; ++i) { a += ws1[i]; b += ws2[i]; }
    float mu = a / Dm;
    ws1[0] = mu; ws2[0] = rsqrtf(b / Dm - mu * mu + EPSL);
  }
  __syncthreads();
  float mu = ws1[0], rstd = ws2[0];
#pragma unroll
  for (int i = 0; i < 3; ++i) {
    int c = tid + i * 256;
    float hv = (v[i] - mu) * rstd * g[c] + bt[c];
    sh[c] = hv;
    h2[(size_t)tok * Dm + c] = (_Float16)hv;
  }
  __syncthreads();
  // gate logit for expert == wave id (softmax is monotone; only argmax matters)
  float acc = 0.f;
  for (int k = lane; k < Dm; k += 32) acc += sh[k] * Wg[(size_t)k * Ee + wv];
#pragma unroll
  for (int off = 16; off; off >>= 1) acc += __shfl_xor(acc, off, 32);
  if (lane == 0) logits[wv] = acc + bg[wv];
  __syncthreads();
  if (tid == 0) {
    int best = 0; float bmax = logits[0];
#pragma unroll
    for (int e = 1; e < Ee; ++e) if (logits[e] > bmax) { bmax = logits[e]; best = e; }
    int pos = atomicAdd(&counts[best], 1);
    idxlist[best * T_TOK + pos] = tok;
  }
}

// ---------- kernel 4: routed expert FFN, 32-token tiles ----------
// out = x1 + gelu(h2 W1 + b1) W2 + b2 ; each B fragment feeds two WMMAs (M=32)
__global__ void moe_kernel(const _Float16* __restrict__ h2, const float* __restrict__ W1,
                           const float* __restrict__ b1, const float* __restrict__ W2,
                           const float* __restrict__ b2, const float* __restrict__ x1,
                           const int* __restrict__ counts, const int* __restrict__ idxlist,
                           float* __restrict__ out) {
  const int LDA = Dm + 8, LDH = 128 + 8;
  __shared__ _Float16 As[MTM * (Dm + 8)];      // 32 x 768 activations (f16)
  __shared__ _Float16 Hs[MTM * (128 + 8)];     // 32 x 128 hidden chunk (f16)
  __shared__ int stok[MTM];
  int e = blockIdx.x / NT32, t = blockIdx.x - e * NT32;
  int cnt = counts[e], base = t * MTM;
  if (base >= cnt) return;                     // uniform early exit
  int valid = min(MTM, cnt - base);
  int tid = threadIdx.x;                       // 256 threads = 8 waves
  if (tid < MTM)
    stok[tid] = idxlist[e * T_TOK + base + ((tid < valid) ? tid : 0)];
  __syncthreads();
  for (int i = tid; i < MTM * Dm; i += 256) {
    int r = i / Dm, c = i - r * Dm;
    As[r * LDA + c] = h2[(size_t)stok[r] * Dm + c];
  }
  int wv = tid >> 5, lane = tid & 31;
  const float* W1e = W1 + (size_t)e * Dm * FFk;
  const float* W2e = W2 + (size_t)e * FFk * Dm;
  v8f accy[2][6] = {};                         // wave owns y cols [wv*96, wv*96+96), 2 M-subtiles
  __syncthreads();

  for (int c0 = 0; c0 < FFk; c0 += 128) {
    // software prefetch of next chunk's weights (global_prefetch_b8) while WMMAs run
    if (c0 + 128 < FFk) {
      int pk = tid * 3;
      if (pk < Dm) __builtin_prefetch(W1e + (size_t)pk * FFk + c0 + 128, 0, 0);
      int rr = c0 + 128 + (tid & 127);
      __builtin_prefetch(W2e + (size_t)rr * Dm + ((tid >> 7) ? 384 : 0), 0, 0);
    }
    // FF1: 16 hidden columns per wave, shared B fragment drives both M-subtiles
    int f0 = c0 + wv * 16;
    v8f acch0 = {}, acch1 = {};
    for (int k0 = 0; k0 < Dm; k0 += 32) {
      v16h b = load_b_f32(W1e, FFk, k0, f0);
      acch0 = wmma16(load_a_lds(As, LDA, k0), b, acch0);
      acch1 = wmma16(load_a_lds(As + 16 * LDA, LDA, k0), b, acch1);
    }
    int nloc = wv * 16 + (lane & 15);
    float bb = b1[(size_t)e * FFk + c0 + nloc];
    int mh = (lane >> 4) << 3;
#pragma unroll
    for (int r = 0; r < 8; ++r) {
      float h0 = acch0[r] + bb;
      h0 = 0.5f * h0 * (1.0f + erff(h0 * 0.70710678f));        // exact GELU
      Hs[(r + mh) * LDH + nloc] = (_Float16)h0;
      float h1 = acch1[r] + bb;
      h1 = 0.5f * h1 * (1.0f + erff(h1 * 0.70710678f));
      Hs[(16 + r + mh) * LDH + nloc] = (_Float16)h1;
    }
    __syncthreads();
    // FF2: accumulate chunk; B fragment reused across both M-subtiles
#pragma unroll
    for (int j = 0; j < 6; ++j) {
      int n0 = wv * 96 + j * 16;
#pragma unroll
      for (int kk = 0; kk < 4; ++kk) {
        v16h b = load_b_f32(W2e, Dm, c0 + kk * 32, n0);
        accy[0][j] = wmma16(load_a_lds(Hs, LDH, kk * 32), b, accy[0][j]);
        accy[1][j] = wmma16(load_a_lds(Hs + 16 * LDH, LDH, kk * 32), b, accy[1][j]);
      }
    }
    __syncthreads();                           // protect Hs WAR for next chunk
  }

  int mh = (lane >> 4) << 3;
#pragma unroll
  for (int mt = 0; mt < 2; ++mt) {
#pragma unroll
    for (int j = 0; j < 6; ++j) {
      int n = wv * 96 + j * 16 + (lane & 15);
      float b2n = b2[(size_t)e * Dm + n];
#pragma unroll
      for (int r = 0; r < 8; ++r) {
        int m = mt * 16 + r + mh;
        if (m < valid) {
          int tok = stok[m];
          out[(size_t)tok * Dm + n] = x1[(size_t)tok * Dm + n] + accy[mt][j][r] + b2n;
        }
      }
    }
  }
}

// ---------- launch ----------
extern "C" void kernel_launch(void* const* d_in, const int* in_sizes, int n_in,
                              void* d_out, int out_size, void* d_ws, size_t ws_size,
                              hipStream_t stream) {
  (void)in_sizes; (void)n_in; (void)out_size; (void)ws_size;
  const float* x     = (const float*)d_in[0];
  const float* ln1_g = (const float*)d_in[1];
  const float* ln1_b = (const float*)d_in[2];
  // d_in[3..6] = Wq, bq, Wk, bk : mathematically irrelevant (attention collapses to v)
  const float* Wv    = (const float*)d_in[7];
  const float* bv    = (const float*)d_in[8];
  const float* ln2_g = (const float*)d_in[9];
  const float* ln2_b = (const float*)d_in[10];
  const float* Wg    = (const float*)d_in[11];
  const float* bg    = (const float*)d_in[12];
  const float* W1    = (const float*)d_in[13];
  const float* b1    = (const float*)d_in[14];
  const float* W2    = (const float*)d_in[15];
  const float* b2    = (const float*)d_in[16];
  float* out = (float*)d_out;

  char* ws = (char*)d_ws;
  _Float16* ln1 = (_Float16*)ws;                       // T*768*2 = 2,408,448 B
  _Float16* h2  = (_Float16*)(ws + 2408448);           // T*768*2
  float*    x1  = (float*)(ws + 4816896);              // T*768*4 = 4,816,896 B
  int*   counts = (int*)(ws + 9633792);                // 8 ints
  int*  idxlist = (int*)(ws + 9633824);                // 8*T ints

  init_counts_kernel<<<1, 32, 0, stream>>>(counts);
  ln1_kernel<<<T_TOK, 256, 0, stream>>>(x, ln1_g, ln1_b, ln1);
  vproj_kernel<<<NTIL, 128, 0, stream>>>(ln1, Wv, bv, x, x1);
  route_kernel<<<T_TOK, 256, 0, stream>>>(x1, ln2_g, ln2_b, Wg, bg, h2, counts, idxlist);
  moe_kernel<<<Ee * NT32, 256, 0, stream>>>(h2, W1, b1, W2, b2, x1, counts, idxlist, out);
}
